// NERCRFClassifier_69535520522486
// MI455X (gfx1250) — matrix operand order repaired and checked by reading, not compiled, not run on hardware
//
#include <hip/hip_runtime.h>
#include <hip/hip_bf16.h>

typedef __attribute__((ext_vector_type(16))) _Float16 v16h;
typedef __attribute__((ext_vector_type(8)))  float    v8f;

#define B_  64
#define T_  512
#define H_  768
#define K_  9
#define NEGC (-10000.0f)

// leaky_relu(x) == max(x, 0.01*x) for finite x (0.01 > 0), which lowers to
// v_pk_mul_f32 + v_pk_max_num_f32 instead of cmp+cndmask pairs.
__device__ __forceinline__ _Float16 lrelu_h(float x) {
  return (_Float16)fmaxf(x, 0.01f * x);
}

__device__ __forceinline__ float wave_sum32(float v) {
#pragma unroll
  for (int o = 16; o > 0; o >>= 1) v += __shfl_xor(v, o, 32);
  return v;
}

// BIO constraint: labels = [O, B-E0, I-E0, B-E1, I-E1, B-E2, I-E2, B-E3, I-E3]
__device__ __forceinline__ bool bio_allowed(int i, int j) {
  if (j == 0) return true;            // -> 'O'
  if (j & 1)  return true;            // -> 'B-Ek' (odd indices)
  return (i == j) || (i == j - 1);    // -> 'I-Ek' only from B-Ek / I-Ek
}

// ---------------------------------------------------------------------------
// Kernel 1: logits = log_softmax(leaky_relu(emb) @ W + b)  via WMMA f16->f32
// Grid: 256 blocks x 256 threads (8 waves); each wave: 16 rows, 24 K-chunks.
// ---------------------------------------------------------------------------
__global__ void __launch_bounds__(256)
proj_logsoftmax_kernel(const float* __restrict__ emb,
                       const float* __restrict__ W,
                       const float* __restrict__ bias,
                       float* __restrict__ logits) {
  __shared__ v16h Bsh[24 * 32];   // 24 KB: B fragments for all 24 K-chunks
  const int tid = threadIdx.x;

  // Build B fragments once per block.
  // B tile is 32x16 (KxN): lanes 0-15 carry K=0-7,16-23, lanes 16-31 carry
  // K=8-15,24-31; element e => K = 8*half + (e&7) + 16*(e>>3), N = lane&15.
  for (int e2 = tid; e2 < 24 * 32; e2 += 256) {
    const int c  = e2 >> 5;
    const int L  = e2 & 31;
    const int n  = L & 15;
    const int kb = (L >> 4) << 3;
    v16h f;
#pragma unroll
    for (int el = 0; el < 16; ++el) {
      const int kg = c * 32 + kb + (el & 7) + ((el >> 3) << 4);  // global H index
      const float v = (n < K_) ? W[kg * K_ + n] : 0.0f;
      f[el] = (_Float16)v;
    }
    Bsh[e2] = f;
  }
  __syncthreads();

  const int wave = tid >> 5;
  const int lane = tid & 31;
  const int base = blockIdx.x * 128 + wave * 16;   // first row of this tile
  const int mrow = lane & 15;
  const int hsel = lane >> 4;     // 0: K 0-7 / 16-23 ; 1: K 8-15 / 24-31
  const int kofA = hsel << 3;

  const float* rowp = emb + (size_t)(base + mrow) * H_;

  v8f acc = {0.f, 0.f, 0.f, 0.f, 0.f, 0.f, 0.f, 0.f};
#pragma unroll 4
  for (int c = 0; c < 24; ++c) {
    const float* p = rowp + c * 32 + kofA;
    float4 f0 = *(const float4*)(p);
    float4 f1 = *(const float4*)(p + 4);
    float4 f2 = *(const float4*)(p + 16);
    float4 f3 = *(const float4*)(p + 20);
    v16h a;
    a[0]  = lrelu_h(f0.x); a[1]  = lrelu_h(f0.y); a[2]  = lrelu_h(f0.z); a[3]  = lrelu_h(f0.w);
    a[4]  = lrelu_h(f1.x); a[5]  = lrelu_h(f1.y); a[6]  = lrelu_h(f1.z); a[7]  = lrelu_h(f1.w);
    a[8]  = lrelu_h(f2.x); a[9]  = lrelu_h(f2.y); a[10] = lrelu_h(f2.z); a[11] = lrelu_h(f2.w);
    a[12] = lrelu_h(f3.x); a[13] = lrelu_h(f3.y); a[14] = lrelu_h(f3.z); a[15] = lrelu_h(f3.w);
    v16h bf = Bsh[c * 32 + lane];
    acc = __builtin_amdgcn_wmma_f32_16x16x32_f16(
        /*neg_a=*/false, a, /*neg_b=*/false, bf,
        /*c_mod=*/(short)0, acc, /*reuse_a=*/false, /*reuse_b=*/false);
  }

  // Epilogue: C/D layout => lane gives N=lane&15, VGPR r gives M=r+8*hsel.
  // log_softmax over N (only n<9 valid) via 16-lane shfl reductions.
  const int n    = lane & 15;
  const int mofs = hsel << 3;
  const float bn = (n < K_) ? bias[n] : 0.0f;
#pragma unroll
  for (int r = 0; r < 8; ++r) {
    float v = (n < K_) ? (acc[r] + bn) : -1.0e30f;
    float mx = v;
#pragma unroll
    for (int o = 1; o < 16; o <<= 1) mx = fmaxf(mx, __shfl_xor(mx, o, 32));
    float s = (n < K_) ? expf(v - mx) : 0.0f;
#pragma unroll
    for (int o = 1; o < 16; o <<= 1) s += __shfl_xor(s, o, 32);
    const float out = v - (mx + logf(s));
    if (n < K_) logits[(size_t)(base + mofs + r) * K_ + n] = out;
  }
}

// ---------------------------------------------------------------------------
// Kernel 2: CRF forward (logZ) + gold score; one wave per batch element.
// partial[b] = logZ_b - gold_b
// ---------------------------------------------------------------------------
__global__ void __launch_bounds__(32)
crf_forward_kernel(const float* __restrict__ logits,
                   const int* __restrict__ mask,
                   const int* __restrict__ label,
                   const float* __restrict__ trans,
                   const float* __restrict__ start_t,
                   const float* __restrict__ end_t,
                   float* __restrict__ partial) {
  const int b = blockIdx.x;
  const int j = threadIdx.x;
  const size_t bT = (size_t)b * T_;

  float tr[K_];
#pragma unroll
  for (int i = 0; i < K_; ++i) tr[i] = (j < K_) ? trans[i * K_ + j] : 0.0f;

  float alpha = (j < K_) ? (start_t[j] + logits[bT * K_ + j]) : -1.0e30f;

  for (int t = 1; t < T_; ++t) {
    const int   mm = mask[bT + t];
    const float e  = (j < K_) ? logits[(bT + t) * K_ + j] : 0.0f;
    float inner[K_];
    float mx = -1.0e30f;
#pragma unroll
    for (int i = 0; i < K_; ++i) {
      const float ai = __shfl(alpha, i, 32);
      inner[i] = ai + tr[i];
      mx = fmaxf(mx, inner[i]);
    }
    float s = 0.0f;
#pragma unroll
    for (int i = 0; i < K_; ++i) s += expf(inner[i] - mx);
    const float nw = mx + logf(s) + e;
    if (mm && (j < K_)) alpha = nw;
  }

  // logZ = logsumexp_j(alpha_j + end_t[j])
  float val = (j < K_) ? (alpha + end_t[j]) : -1.0e30f;
  float mx = val;
#pragma unroll
  for (int o = 16; o > 0; o >>= 1) mx = fmaxf(mx, __shfl_xor(mx, o, 32));
  float s = (j < K_) ? expf(val - mx) : 0.0f;
  s = wave_sum32(s);
  const float logZ = mx + logf(s);

  // Gold score: lanes stride over t.
  float ge = 0.0f, gt = 0.0f;
  float len = 0.0f;
  for (int t = j; t < T_; t += 32) {
    const int mm  = mask[bT + t];
    const int lab = label[bT + t];
    if (mm) {
      ge  += logits[(bT + t) * K_ + lab];
      len += 1.0f;
      if (t >= 1) gt += trans[label[bT + t - 1] * K_ + lab];
    }
  }
  ge  = wave_sum32(ge);
  gt  = wave_sum32(gt);
  len = wave_sum32(len);

  if (j == 0) {
    const int length   = (int)len;
    const int last_tag = label[bT + length - 1];
    const float gold = start_t[label[bT]] + ge + gt + end_t[last_tag];
    partial[b] = logZ - gold;
  }
}

// ---------------------------------------------------------------------------
// Kernel 3: loss = sum_b(logZ_b - gold_b) / B
// ---------------------------------------------------------------------------
__global__ void __launch_bounds__(32)
loss_reduce_kernel(const float* __restrict__ partial, float* __restrict__ outf) {
  const int j = threadIdx.x;
  float v = partial[j] + partial[j + 32];
  v = wave_sum32(v);
  if (j == 0) outf[0] = v / (float)B_;
}

// ---------------------------------------------------------------------------
// Kernel 4: BIO-constrained Viterbi + backtrace; one wave per batch element.
// Backpointers staged in LDS (512 x 9 bytes), lane 0 backtraces.
// ---------------------------------------------------------------------------
__global__ void __launch_bounds__(32)
viterbi_kernel(const float* __restrict__ logits,
               const int* __restrict__ mask,
               const float* __restrict__ trans,
               const float* __restrict__ start_t,
               const float* __restrict__ end_t,
               int* __restrict__ paths) {
  __shared__ unsigned char bp[T_ * K_];
  const int b = blockIdx.x;
  const int j = threadIdx.x;
  const size_t bT = (size_t)b * T_;

  float trc[K_];
#pragma unroll
  for (int i = 0; i < K_; ++i)
    trc[i] = (j < K_) ? (bio_allowed(i, j) ? trans[i * K_ + j] : NEGC) : 0.0f;

  const float st = (j < K_) ? (((j == 0) || (j & 1)) ? start_t[j] : NEGC) : 0.0f;
  float score = (j < K_) ? (st + logits[bT * K_ + j]) : -1.0e30f;

  for (int t = 1; t < T_; ++t) {
    const int   mm = mask[bT + t];
    const float e  = (j < K_) ? logits[(bT + t) * K_ + j] : 0.0f;
    float best = -3.0e38f;
    int bi = 0;
#pragma unroll
    for (int i = 0; i < K_; ++i) {
      const float v = __shfl(score, i, 32) + trc[i];
      if (v > best) { best = v; bi = i; }   // strict '>' => first-max like argmax
    }
    if (j < K_) bp[t * K_ + j] = (unsigned char)(mm ? bi : j);
    if (mm && (j < K_)) score = best + e;
  }

  // last = argmax_j(score_j + end_t[j]) (first max)
  const float val = (j < K_) ? (score + end_t[j]) : -1.0e30f;
  float best = -3.0e38f;
  int tag = 0;
  for (int q = 0; q < K_; ++q) {
    const float v = __shfl(val, q, 32);
    if (v > best) { best = v; tag = q; }
  }
  __syncthreads();

  if (j == 0) {
    int tg = tag;
    for (int t = T_ - 1; t >= 1; --t) {
      paths[bT + t] = mask[bT + t] ? tg : 0;
      tg = bp[t * K_ + tg];
    }
    paths[bT] = mask[bT] ? tg : 0;
  }
}

// ---------------------------------------------------------------------------
extern "C" void kernel_launch(void* const* d_in, const int* in_sizes, int n_in,
                              void* d_out, int out_size, void* d_ws, size_t ws_size,
                              hipStream_t stream) {
  (void)in_sizes; (void)n_in; (void)out_size; (void)ws_size;

  const float* emb     = (const float*)d_in[0];   // (B,T,H) f32
  const int*   mask    = (const int*)  d_in[1];   // (B,T) i32
  const int*   label   = (const int*)  d_in[2];   // (B,T) i32
  const float* W       = (const float*)d_in[3];   // (H,K) f32
  const float* bias    = (const float*)d_in[4];   // (K,)  f32
  const float* trans   = (const float*)d_in[5];   // (K,K) f32
  const float* start_t = (const float*)d_in[6];   // (K,)  f32
  const float* end_t   = (const float*)d_in[7];   // (K,)  f32

  float* outf  = (float*)d_out;        // [0] = loss
  int*   paths = (int*)d_out + 1;      // (B,T) int32 best paths

  float* logits_ws = (float*)d_ws;                       // B*T*K f32
  float* partial   = logits_ws + (size_t)B_ * T_ * K_;   // B f32

  // 1) projection + log_softmax (WMMA)
  proj_logsoftmax_kernel<<<256, 256, 0, stream>>>(emb, W, bias, logits_ws);
  // 2) CRF forward + gold
  crf_forward_kernel<<<B_, 32, 0, stream>>>(logits_ws, mask, label, trans,
                                            start_t, end_t, partial);
  // 3) scalar loss
  loss_reduce_kernel<<<1, 32, 0, stream>>>(partial, outf);
  // 4) constrained Viterbi + backtrace
  viterbi_kernel<<<B_, 32, 0, stream>>>(logits_ws, mask, trans,
                                        start_t, end_t, paths);
}
